// BMEDLSTM_46359876993047
// MI455X (gfx1250) — matrix operand order repaired and checked by reading, not compiled, not run on hardware
//
#include <hip/hip_runtime.h>
#include <hip/hip_bf16.h>
#include <math.h>

typedef _Float16 v16h __attribute__((ext_vector_type(16)));
typedef _Float16 v8h  __attribute__((ext_vector_type(8)));
typedef float    v8f  __attribute__((ext_vector_type(8)));

constexpr int cB  = 128;   // batch
constexpr int cXS = 64;    // x size
constexpr int cSS = 16;    // s size
constexpr int cH  = 512;   // hidden
constexpr int cG  = 2048;  // 4*H
constexpr int cK0 = 608;   // 64 x + 16 s + 16 pad + 512 h0
constexpr int cK1 = 1024;  // 512 h0 + 512 h1
constexpr int cT  = 201;   // int(20/0.1)+1
constexpr float cDT = 0.1f;

__device__ __forceinline__ float sigm_(float v) { return 1.0f / (1.0f + expf(-v)); }

// ---------------------------------------------------------------------------
// Prep: f16 weight repack (K-concatenated, row-major [N x K]), fused biases,
// state init, static activation regions, cal_times.
// ---------------------------------------------------------------------------
__global__ void prep_kernel(
    const float* __restrict__ x,    const float* __restrict__ s0,
    const float* __restrict__ Wih0, const float* __restrict__ Whh0,
    const float* __restrict__ bih0, const float* __restrict__ bhh0,
    const float* __restrict__ Wih1, const float* __restrict__ Whh1,
    const float* __restrict__ bih1, const float* __restrict__ bhh1,
    const float* __restrict__ Wfc,
    _Float16* __restrict__ W0c,  _Float16* __restrict__ W1c, _Float16* __restrict__ Wfc16,
    float* __restrict__ bias0,   float* __restrict__ bias1,
    float* __restrict__ c0,      float* __restrict__ c1,     float* __restrict__ sb0,
    _Float16* __restrict__ Ac0,  _Float16* __restrict__ Ac1,
    _Float16* __restrict__ A1c0, _Float16* __restrict__ A1c1,
    float* __restrict__ cal)
{
    int i = blockIdx.x * blockDim.x + threadIdx.x;

    const int nW0 = cG * cK0;
    const int nW1 = cG * cK1;
    const int nWf = cSS * cH;
    const int nBi = cG;
    const int nC  = cB * cH;
    const int nS  = cB * cSS;
    const int nAc = 2 * cB * cK0;
    const int nA1 = 2 * cB * cK1;

    if (i < nW0) {                         // W0cat[n,k] = [W_ih0(80) | pad(16) | W_hh0(512)]
        int n = i / cK0, k = i % cK0;
        float v = 0.0f;
        if (k < 80)       v = Wih0[n * 80 + k];
        else if (k >= 96) v = Whh0[n * cH + (k - 96)];
        W0c[i] = (_Float16)v;
        return;
    }
    i -= nW0;
    if (i < nW1) {                         // W1cat[n,k] = [W_ih1(512) | W_hh1(512)]
        int n = i / cK1, k = i % cK1;
        float v = (k < cH) ? Wih1[n * cH + k] : Whh1[n * cH + (k - cH)];
        W1c[i] = (_Float16)v;
        return;
    }
    i -= nW1;
    if (i < nWf) { Wfc16[i] = (_Float16)Wfc[i]; return; }
    i -= nWf;
    if (i < nBi) { bias0[i] = bih0[i] + bhh0[i]; return; }
    i -= nBi;
    if (i < nBi) { bias1[i] = bih1[i] + bhh1[i]; return; }
    i -= nBi;
    if (i < nC)  { c0[i] = 0.0f; return; }
    i -= nC;
    if (i < nC)  { c1[i] = 0.0f; return; }
    i -= nC;
    if (i < nS)  { sb0[i] = s0[i]; return; }   // s(0) = s0
    i -= nS;
    if (i < nAc) {                         // Acat0 both parities: [x | s(parity0)=s0 | pad | h0=0]
        int pp = i / (cB * cK0);
        int rem = i - pp * (cB * cK0);
        int b = rem / cK0, k = rem - b * cK0;
        float v = 0.0f;
        if (k < cXS)                  v = x[b * cXS + k];
        else if (k < 80 && pp == 0)   v = s0[b * cSS + (k - cXS)];
        (pp ? Ac1 : Ac0)[b * cK0 + k] = (_Float16)v;
        return;
    }
    i -= nAc;
    if (i < nA1) {                         // A1cat both parities = 0 (h0(0)=h1(0)=0)
        int pp = i / (cB * cK1);
        int rem = i - pp * (cB * cK1);
        (pp ? A1c1 : A1c0)[rem] = (_Float16)0.0f;
        return;
    }
    i -= nA1;
    if (i < cT)  { cal[i] = (float)i * cDT; return; }
}

// ---------------------------------------------------------------------------
// Fused step-front kernel (per step t):
//   (a) if dofc: out(t-1) = h1(t) @ Wfc^T + bfc (WMMA, per-wave rows);
//       s(t) = s_in + DT*out; scatter outputs[t-1], res[t-1], s_out;
//       write s(t) f16 into this wave's own A rows (cols 64..79 of Acat0[p]).
//   (b) layer-0 LSTM: gates = A[128xK0] @ W0cat[2048xK0]^T + bias0; cell update;
//       h0(t+1) -> A1cat[p] cols 0..511 and Acat0[1-p] cols 96..607.
// grid = (32), block = 256 (8 waves; wave w owns batch rows 16w..16w+15 and
// this block's 16-wide hidden slice j0 = 16*blockIdx.x).
// Redundant FC stores across the 32 blocks write identical values (benign).
// ---------------------------------------------------------------------------
__global__ void lstm_front_kernel(
    _Float16* __restrict__ Ac,           // Acat0[p]: A source; also s f16 dest (cols 64..79)
    const _Float16* __restrict__ A1cur,  // A1cat[p]: h1(t) at col 512 (read)
    const _Float16* __restrict__ W0,     // [2048 x 608] f16
    const _Float16* __restrict__ Wfc16,  // [16 x 512] f16
    const float* __restrict__ bias0, const float* __restrict__ bfc,
    float* __restrict__ cstate,          // c0 [128 x 512] f32
    _Float16* __restrict__ h1out,        // A1cat[p] cols 0..511
    _Float16* __restrict__ h2out,        // Acat0[1-p] cols 96..607
    const float* __restrict__ s_in, float* __restrict__ s_out,
    float* __restrict__ outputs, float* __restrict__ res,
    int tprev, int dofc)
{
    const int lane = threadIdx.x & 31;
    const int wave = threadIdx.x >> 5;   // 0..7 == M tile
    const int j0   = blockIdx.x << 4;    // hidden slice base
    const int r    = lane & 15;
    const int kh   = lane >> 4;
    const int row  = (wave << 4) + r;

    // ---- (a) FC + Euler update for this wave's 16 batch rows ----
    if (dofc) {
        const _Float16* Arow = A1cur + (size_t)row * cK1 + cH;   // h1(t)
        const _Float16* Brow = Wfc16 + (size_t)r * cH;
        v8f acc = {};
        for (int k0 = 0; k0 < cH; k0 += 32) {
            const _Float16* ap = Arow + k0 + (kh << 4);
            v8h alo = *(const v8h*)(ap);
            v8h ahi = *(const v8h*)(ap + 8);
            v16h af = __builtin_shufflevector(alo, ahi,
                        0,1,2,3,4,5,6,7,8,9,10,11,12,13,14,15);
            v16h bf = *(const v16h*)(Brow + k0 + (kh << 4));
            acc = __builtin_amdgcn_wmma_f32_16x16x32_f16(false, af, false, bf,
                                                         (short)0, acc, false, false);
        }
        const float bb = bfc[r];
#pragma unroll
        for (int v = 0; v < 8; ++v) {
            const int brow = (wave << 4) + (kh << 3) + v;
            float o  = acc[v] + bb;
            int   si = brow * cSS + r;
            float sv = s_in[si] + cDT * o;
            s_out[si] = sv;
            Ac[brow * cK0 + cXS + r] = (_Float16)sv;   // own rows only
            outputs[((size_t)brow * cT + tprev) * cSS + r] = sv;
            res    [((size_t)brow * cT + tprev) * cSS + r] = o;
        }
        // Ensure the s f16 stores above are visible to this wave's A loads below
        // (ISA keeps same-wave same-address store->load in order; the wait + memory
        // clobber also pins the compiler ordering).
        asm volatile("s_wait_storecnt 0x0" ::: "memory");
    }

    // ---- (b) layer-0 gates + cell ----
    const _Float16* Arow = Ac + (size_t)row * cK0;
    const _Float16* B0 = W0 + (size_t)(j0 + r) * cK0;     // gate i
    const _Float16* B1 = B0 + (size_t)cH * cK0;           // gate f
    const _Float16* B2 = B1 + (size_t)cH * cK0;           // gate g
    const _Float16* B3 = B2 + (size_t)cH * cK0;           // gate o

    v8f acc0 = {}, acc1 = {}, acc2 = {}, acc3 = {};
    for (int k0 = 0; k0 < cK0; k0 += 32) {
        const _Float16* ap = Arow + k0 + (kh << 4);
        v8h alo = *(const v8h*)(ap);
        v8h ahi = *(const v8h*)(ap + 8);
        v16h af = __builtin_shufflevector(alo, ahi,
                    0,1,2,3,4,5,6,7,8,9,10,11,12,13,14,15);
        const int bo = k0 + (kh << 4);
        v16h b0 = *(const v16h*)(B0 + bo);
        v16h b1 = *(const v16h*)(B1 + bo);
        v16h b2 = *(const v16h*)(B2 + bo);
        v16h b3 = *(const v16h*)(B3 + bo);
        acc0 = __builtin_amdgcn_wmma_f32_16x16x32_f16(false, af, false, b0, (short)0, acc0, false, false);
        acc1 = __builtin_amdgcn_wmma_f32_16x16x32_f16(false, af, false, b1, (short)0, acc1, false, false);
        acc2 = __builtin_amdgcn_wmma_f32_16x16x32_f16(false, af, false, b2, (short)0, acc2, false, false);
        acc3 = __builtin_amdgcn_wmma_f32_16x16x32_f16(false, af, false, b3, (short)0, acc3, false, false);
    }

    const int j = j0 + r;
    const float bvi = bias0[j];
    const float bvf = bias0[cH + j];
    const float bvg = bias0[2 * cH + j];
    const float bvo = bias0[3 * cH + j];
#pragma unroll
    for (int v = 0; v < 8; ++v) {
        const int brow = (wave << 4) + (kh << 3) + v;
        float ig = sigm_(acc0[v] + bvi);
        float fg = sigm_(acc1[v] + bvf);
        float gt = tanhf(acc2[v] + bvg);
        float og = sigm_(acc3[v] + bvo);
        float cc = fg * cstate[brow * cH + j] + ig * gt;
        cstate[brow * cH + j] = cc;
        _Float16 hh = (_Float16)(og * tanhf(cc));
        h1out[brow * cK1 + j] = hh;
        h2out[brow * cK0 + 96 + j] = hh;
    }
}

// ---------------------------------------------------------------------------
// Layer-1 LSTM: gates = A1cat[p][128x1024] @ W1cat^T + bias1; cell update;
// h1(t+1) -> A1cat[1-p] cols 512..1023.  grid = (32, 2), block = 128.
// ---------------------------------------------------------------------------
__global__ void lstm_layer1_kernel(
    const _Float16* __restrict__ A,  const _Float16* __restrict__ W,
    const float* __restrict__ bias,  float* __restrict__ cstate,
    _Float16* __restrict__ hout)     // A1cat[1-p]; h1 goes to col 512
{
    const int lane = threadIdx.x & 31;
    const int wave = threadIdx.x >> 5;
    const int mt   = blockIdx.y * 4 + wave;   // 0..7
    const int j0   = blockIdx.x << 4;
    const int r    = lane & 15;
    const int kh   = lane >> 4;
    const int row  = (mt << 4) + r;

    const _Float16* Arow = A + (size_t)row * cK1;
    const _Float16* B0 = W + (size_t)(j0 + r) * cK1;
    const _Float16* B1 = B0 + (size_t)cH * cK1;
    const _Float16* B2 = B1 + (size_t)cH * cK1;
    const _Float16* B3 = B2 + (size_t)cH * cK1;

    v8f acc0 = {}, acc1 = {}, acc2 = {}, acc3 = {};
    for (int k0 = 0; k0 < cK1; k0 += 32) {
        const _Float16* ap = Arow + k0 + (kh << 4);
        v8h alo = *(const v8h*)(ap);
        v8h ahi = *(const v8h*)(ap + 8);
        v16h af = __builtin_shufflevector(alo, ahi,
                    0,1,2,3,4,5,6,7,8,9,10,11,12,13,14,15);
        const int bo = k0 + (kh << 4);
        v16h b0 = *(const v16h*)(B0 + bo);
        v16h b1 = *(const v16h*)(B1 + bo);
        v16h b2 = *(const v16h*)(B2 + bo);
        v16h b3 = *(const v16h*)(B3 + bo);
        acc0 = __builtin_amdgcn_wmma_f32_16x16x32_f16(false, af, false, b0, (short)0, acc0, false, false);
        acc1 = __builtin_amdgcn_wmma_f32_16x16x32_f16(false, af, false, b1, (short)0, acc1, false, false);
        acc2 = __builtin_amdgcn_wmma_f32_16x16x32_f16(false, af, false, b2, (short)0, acc2, false, false);
        acc3 = __builtin_amdgcn_wmma_f32_16x16x32_f16(false, af, false, b3, (short)0, acc3, false, false);
    }

    const int j = j0 + r;
    const float bvi = bias[j];
    const float bvf = bias[cH + j];
    const float bvg = bias[2 * cH + j];
    const float bvo = bias[3 * cH + j];
#pragma unroll
    for (int v = 0; v < 8; ++v) {
        const int brow = (mt << 4) + (kh << 3) + v;
        float ig = sigm_(acc0[v] + bvi);
        float fg = sigm_(acc1[v] + bvf);
        float gt = tanhf(acc2[v] + bvg);
        float og = sigm_(acc3[v] + bvo);
        float cc = fg * cstate[brow * cH + j] + ig * gt;
        cstate[brow * cH + j] = cc;
        hout[brow * cK1 + cH + j] = (_Float16)(og * tanhf(cc));
    }
}

// ---------------------------------------------------------------------------
// Tail FC (final step only): out(T-1) = h1(T) @ Wfc^T + bfc; s += DT*out;
// scatter outputs[T-1], res[T-1]. One workgroup, 8 waves.
// ---------------------------------------------------------------------------
__global__ void fc_tail_kernel(
    const _Float16* __restrict__ A1n,   // A1cat holding h1(T) at col 512
    const _Float16* __restrict__ Wfc16,
    const float* __restrict__ bfc,
    const float* __restrict__ s_in, float* __restrict__ s_out,
    float* __restrict__ outputs, float* __restrict__ res, int t)
{
    const int lane = threadIdx.x & 31;
    const int wave = threadIdx.x >> 5;
    const int r    = lane & 15;
    const int kh   = lane >> 4;
    const int row  = (wave << 4) + r;

    const _Float16* Arow = A1n + (size_t)row * cK1 + cH;
    const _Float16* Brow = Wfc16 + (size_t)r * cH;

    v8f acc = {};
    for (int k0 = 0; k0 < cH; k0 += 32) {
        const _Float16* ap = Arow + k0 + (kh << 4);
        v8h alo = *(const v8h*)(ap);
        v8h ahi = *(const v8h*)(ap + 8);
        v16h af = __builtin_shufflevector(alo, ahi,
                    0,1,2,3,4,5,6,7,8,9,10,11,12,13,14,15);
        v16h bf = *(const v16h*)(Brow + k0 + (kh << 4));
        acc = __builtin_amdgcn_wmma_f32_16x16x32_f16(false, af, false, bf,
                                                     (short)0, acc, false, false);
    }

    const float bb = bfc[r];
#pragma unroll
    for (int v = 0; v < 8; ++v) {
        const int brow = (wave << 4) + (kh << 3) + v;
        float o  = acc[v] + bb;
        int   si = brow * cSS + r;
        float sv = s_in[si] + cDT * o;
        s_out[si] = sv;
        outputs[((size_t)brow * cT + t) * cSS + r] = sv;
        res    [((size_t)brow * cT + t) * cSS + r] = o;
    }
}

// ---------------------------------------------------------------------------
extern "C" void kernel_launch(void* const* d_in, const int* in_sizes, int n_in,
                              void* d_out, int out_size, void* d_ws, size_t ws_size,
                              hipStream_t stream) {
    const float* x    = (const float*)d_in[0];
    const float* s0   = (const float*)d_in[1];
    const float* Wih0 = (const float*)d_in[2];
    const float* Whh0 = (const float*)d_in[3];
    const float* bih0 = (const float*)d_in[4];
    const float* bhh0 = (const float*)d_in[5];
    const float* Wih1 = (const float*)d_in[6];
    const float* Whh1 = (const float*)d_in[7];
    const float* bih1 = (const float*)d_in[8];
    const float* bhh1 = (const float*)d_in[9];
    const float* Wfc  = (const float*)d_in[10];
    const float* bfc  = (const float*)d_in[11];
    // d_in[12] = max_time (=20) -> T hardcoded to 201 per reference constants.

    uint8_t* ws = (uint8_t*)d_ws;
    size_t off = 0;
    auto carve = [&](size_t bytes) -> void* {
        void* p = ws + off;
        off = (off + bytes + 255) & ~(size_t)255;
        return p;
    };
    _Float16* W0c   = (_Float16*)carve((size_t)cG * cK0 * 2);
    _Float16* W1c   = (_Float16*)carve((size_t)cG * cK1 * 2);
    _Float16* Wfc16 = (_Float16*)carve((size_t)cSS * cH * 2);
    float*    bias0 = (float*)carve((size_t)cG * 4);
    float*    bias1 = (float*)carve((size_t)cG * 4);
    float*    c0    = (float*)carve((size_t)cB * cH * 4);
    float*    c1    = (float*)carve((size_t)cB * cH * 4);
    float*    sb[2];
    sb[0] = (float*)carve((size_t)cB * cSS * 4);
    sb[1] = (float*)carve((size_t)cB * cSS * 4);
    _Float16* Ac[2];
    Ac[0] = (_Float16*)carve((size_t)cB * cK0 * 2);
    Ac[1] = (_Float16*)carve((size_t)cB * cK0 * 2);
    _Float16* A1[2];
    A1[0] = (_Float16*)carve((size_t)cB * cK1 * 2);
    A1[1] = (_Float16*)carve((size_t)cB * cK1 * 2);

    float* outs = (float*)d_out;                       // [B, T, SS]
    float* cal  = outs + (size_t)cB * cT * cSS;        // [T]
    float* res  = cal + cT;                            // [B, T, SS]

    const int nPrep = cG * cK0 + cG * cK1 + cSS * cH + 2 * cG + 2 * cB * cH
                    + cB * cSS + 2 * cB * cK0 + 2 * cB * cK1 + cT;
    prep_kernel<<<(nPrep + 255) / 256, 256, 0, stream>>>(
        x, s0, Wih0, Whh0, bih0, bhh0, Wih1, Whh1, bih1, bhh1, Wfc,
        W0c, W1c, Wfc16, bias0, bias1, c0, c1, sb[0],
        Ac[0], Ac[1], A1[0], A1[1], cal);

    for (int t = 0; t < cT; ++t) {
        const int p = t & 1;
        // Fused FC(t-1) + layer 0(t):
        //  reads  Acat0[p] (x | s | pad | h0(t)), A1cat[p] col512 (h1(t)),
        //  writes s(t)=sb[t&1] (from sb[(t-1)&1]), outputs/res[t-1],
        //         h0(t+1) -> A1cat[p] cols 0.. and Acat0[1-p] cols 96..
        lstm_front_kernel<<<dim3(cH / 16), 256, 0, stream>>>(
            Ac[p], A1[p], W0c, Wfc16, bias0, bfc, c0,
            A1[p], Ac[1 - p],
            sb[(t + 1) & 1], sb[t & 1],
            outs, res, t - 1, (t > 0) ? 1 : 0);
        // Layer 1(t): reads A1cat[p] = [h0(t+1) | h1(t)];
        //  writes h1(t+1) -> A1cat[1-p] cols 512..1023.
        lstm_layer1_kernel<<<dim3(cH / 16, 2), 128, 0, stream>>>(
            A1[p], W1c, bias1, c1, A1[1 - p]);
    }
    // Final FC: out(200) from h1(201) in A1cat[1] (since p(200)=0 wrote 1-p=1).
    fc_tail_kernel<<<1, 256, 0, stream>>>(
        A1[1], Wfc16, bfc, sb[0], sb[1], outs, res, cT - 1);

    (void)in_sizes; (void)n_in; (void)out_size; (void)ws_size;
}
